// DGCNN_7885559955735
// MI455X (gfx1250) — compile-verified, matrix operand order checked by reading, not run on hardware
//
#include <hip/hip_runtime.h>
#include <hip/hip_bf16.h>

typedef __bf16 bf16_t;
typedef __attribute__((ext_vector_type(8)))  __bf16 bf16x8;
typedef __attribute__((ext_vector_type(16))) __bf16 v16bf;
typedef __attribute__((ext_vector_type(8)))  float  v8f;

#define NPTS 2048
#define KNN  20

// ---------------------------------------------------------------- utilities
__device__ __forceinline__ unsigned long long shflxor64(unsigned long long v, int m) {
    unsigned int lo = (unsigned int)v;
    unsigned int hi = (unsigned int)(v >> 32);
    lo = __shfl_xor(lo, m, 32);
    hi = __shfl_xor(hi, m, 32);
    return ((unsigned long long)hi << 32) | lo;
}

// CDNA5 async Memory->LDS copy (ASYNCcnt-tracked); per-lane 16B transfer.
// %0 = LDS byte address (wave-relative), %1 = 64-bit global address.
__device__ __forceinline__ void async_copy_b128(unsigned int lds_off, const void* gaddr) {
    asm volatile("global_load_async_to_lds_b128 %0, %1, off"
                 :: "v"(lds_off), "v"(gaddr) : "memory");
}
__device__ __forceinline__ void wait_asynccnt0() {
    asm volatile("s_wait_asynccnt 0" ::: "memory");
}

__global__ void zerof_kernel(float* p, int n) {
    int t = blockIdx.x * blockDim.x + threadIdx.x;
    if (t < n) p[t] = 0.0f;
}

// Convert f32 weights (Cout x Cin, row major) to bf16 padded (Cout x Cpad).
__global__ void wconv_kernel(const float* __restrict__ W, bf16_t* __restrict__ Wb,
                             int Cout, int Cin, int Cpad) {
    int t = blockIdx.x * blockDim.x + threadIdx.x;
    if (t >= Cout * Cpad) return;
    int o = t / Cpad, c = t % Cpad;
    Wb[t] = (bf16_t)(c < Cin ? W[o * Cin + c] : 0.0f);
}

// ---------------------------------------------------------------- kNN
// One wave per query point; distances cached in LDS; 20 argmin passes with
// packed (dist_bits, idx) keys reduced via wave32 shfl-xor.
__global__ __launch_bounds__(128) void knn_kernel(const float* __restrict__ x,
                                                  int* __restrict__ idx) {
    __shared__ float sdist[4 * NPTS];   // 32 KB: 4 waves/block
    int t = threadIdx.x;
    int lane = t & 31;
    int wv = t >> 5;
    int q = blockIdx.x * 4 + wv;        // global query id = b*N + n
    int b = q >> 11;                    // N = 2048
    int n = q & (NPTS - 1);

    const float* xb = x + (size_t)b * 3 * NPTS;
    float qx = xb[n], qy = xb[NPTS + n], qz = xb[2 * NPTS + n];

    for (int m = lane; m < NPTS; m += 32) {
        float dx = xb[m] - qx;
        float dy = xb[NPTS + m] - qy;
        float dz = xb[2 * NPTS + m] - qz;
        sdist[wv * NPTS + m] = dx * dx + dy * dy + dz * dz;
    }
    __syncthreads();

    unsigned long long prev = 0ull;
    for (int it = 0; it < KNN; ++it) {
        unsigned long long best = ~0ull;
        for (int m = lane; m < NPTS; m += 32) {
            unsigned int fb = __float_as_uint(sdist[wv * NPTS + m]);
            unsigned long long key = ((unsigned long long)fb << 32) | (unsigned int)m;
            if (it == 0 || key > prev) best = (key < best) ? key : best;
        }
        for (int o = 16; o > 0; o >>= 1) {
            unsigned long long other = shflxor64(best, o);
            best = (other < best) ? other : best;
        }
        if (lane == 0) idx[q * KNN + it] = (int)(best & 0xFFFFFFFFu);
        prev = best;
    }
}

// ---------------------------------------------------------------- graph features
// Build (P1 x 32) bf16 rows: [nbr-ctr (3), ctr (3), zeros (26)]
__global__ void feat_kernel(const float* __restrict__ x, const int* __restrict__ idx,
                            bf16_t* __restrict__ A, long long total) {
    long long p = (long long)blockIdx.x * blockDim.x + threadIdx.x;
    if (p >= total) return;
    int n = (int)((p / KNN) & (NPTS - 1));
    int b = (int)(p / ((long long)KNN * NPTS));
    int j = idx[p];
    const float* xb = x + (size_t)b * 3 * NPTS;
    float cx = xb[n], cy = xb[NPTS + n], cz = xb[2 * NPTS + n];
    float nx = xb[j], ny = xb[NPTS + j], nz = xb[2 * NPTS + j];
    bf16_t* row = A + p * 32;
    row[0] = (bf16_t)(nx - cx);
    row[1] = (bf16_t)(ny - cy);
    row[2] = (bf16_t)(nz - cz);
    row[3] = (bf16_t)cx;
    row[4] = (bf16_t)cy;
    row[5] = (bf16_t)cz;
    #pragma unroll
    for (int c = 6; c < 32; ++c) row[c] = (bf16_t)0.0f;
}

// ---------------------------------------------------------------- WMMA GEMM + fused BN stats
// Y[P][Cout] = A[P][Cin] * W[Cout][Cin]^T  (bf16 in, f32 acc, bf16 out).
// Block: 8 waves x (16M x 64N) register-blocked tiles; weight slab for the
// block's 64 output channels staged to LDS via async copy (ASYNCcnt), B
// fragments served by ds_load_b128. Per-channel sum/sumsq fused for BN.
__global__ __launch_bounds__(256) void wmma_mm_kernel(const bf16_t* __restrict__ A,
                                                      const bf16_t* __restrict__ W,
                                                      bf16_t* __restrict__ Y,
                                                      float* __restrict__ gsum,
                                                      float* __restrict__ gsq,
                                                      int Cin, int Cout) {
    __shared__ bf16_t Bs[64 * 64];      // 8 KB weight slab (64 cols x K-chunk)
    __shared__ float ssum[64];
    __shared__ float ssq[64];

    int t = threadIdx.x;
    int lane = t & 31;
    int wv = t >> 5;
    if (t < 64) { ssum[t] = 0.0f; ssq[t] = 0.0f; }

    int m_base = (blockIdx.x * 8 + wv) * 16;
    int nblk = blockIdx.y * 64;
    int half = lane >> 4;
    int l15 = lane & 15;
    int mrow = m_base + l15;

    const bf16_t* Ap = A + (size_t)mrow * Cin + half * 8;

    const int KC = (Cin >= 64) ? 64 : Cin;   // K-chunk: 64 (or 32 for layer 1)
    const int lg = (KC == 64) ? 3 : 2;       // log2(KC/8)

    v8f acc[4] = {};

    for (int kc = 0; kc < Cin; kc += KC) {
        // ---- stage 64 x KC weight slab into LDS (async, 16B per lane-op)
        #pragma unroll
        for (int v = 0; v < 2; ++v) {
            if (v * 256 < (KC << 3)) {       // total vec16 transfers = 8*KC
                int i = t + v * 256;
                int col = i >> lg;
                int seg = i & ((1 << lg) - 1);
                const bf16_t* g = W + (size_t)(nblk + col) * Cin + kc + seg * 8;
                unsigned int lo = (unsigned int)(uintptr_t)&Bs[(col << (lg + 3)) + seg * 8];
                async_copy_b128(lo, g);
            }
        }
        wait_asynccnt0();
        __syncthreads();

        if (kc + KC < Cin) __builtin_prefetch(Ap + kc + KC, 0, 1);

        // ---- consume chunk: per 32-K step, 1 A fragment feeds 4 WMMAs
        for (int ks = 0; ks < KC; ks += 32) {
            bf16x8 a0 = *(const bf16x8*)(Ap + kc + ks);
            bf16x8 a1 = *(const bf16x8*)(Ap + kc + ks + 16);
            v16bf av;
            #pragma unroll
            for (int i = 0; i < 8; ++i) { av[i] = a0[i]; av[i + 8] = a1[i]; }

            #pragma unroll
            for (int j = 0; j < 4; ++j) {
                int colLoc = j * 16 + l15;
                int bbase = (colLoc << (lg + 3)) + ks + half * 8;
                bf16x8 b0 = *(const bf16x8*)&Bs[bbase];
                bf16x8 b1 = *(const bf16x8*)&Bs[bbase + 16];
                v16bf bv;
                #pragma unroll
                for (int i = 0; i < 8; ++i) { bv[i] = b0[i]; bv[i + 8] = b1[i]; }
                acc[j] = __builtin_amdgcn_wmma_f32_16x16x32_bf16(false, av, false, bv,
                                                                 (short)0, acc[j], false, false);
            }
        }
        __syncthreads();   // protect Bs before next chunk overwrites it
    }

    // ---- epilogue: store Y (bf16) + fused per-channel BN statistics
    #pragma unroll
    for (int j = 0; j < 4; ++j) {
        float s = 0.0f, q = 0.0f;
        int ncol = nblk + j * 16 + l15;
        #pragma unroll
        for (int r = 0; r < 8; ++r) {
            float vv = acc[j][r];
            s += vv; q += vv * vv;
            int row = m_base + r + 8 * half;
            Y[(size_t)row * Cout + ncol] = (bf16_t)vv;
        }
        // lanes L and L^16 hold partials for the same output channel
        s += __shfl_xor(s, 16, 32);
        q += __shfl_xor(q, 16, 32);
        if (lane < 16) { atomicAdd(&ssum[j * 16 + l15], s); atomicAdd(&ssq[j * 16 + l15], q); }
    }
    __syncthreads();
    if (t < 64) {
        atomicAdd(&gsum[nblk + t], ssum[t]);
        atomicAdd(&gsq[nblk + t], ssq[t]);
    }
}

// ---------------------------------------------------------------- BN + ReLU -> bf16
__global__ void bn_apply_kernel(const bf16_t* __restrict__ Y,
                                const float* __restrict__ gsum, const float* __restrict__ gsq,
                                const float* __restrict__ gamma, const float* __restrict__ beta,
                                bf16_t* __restrict__ Anext, int C, float invCnt, long long total) {
    long long t = (long long)blockIdx.x * blockDim.x + threadIdx.x;
    if (t >= total) return;
    int c = (int)(t % C);
    float mean = gsum[c] * invCnt;
    float var  = gsq[c] * invCnt - mean * mean;
    float sc = gamma[c] * rsqrtf(var + 1e-5f);
    float sh = beta[c] - mean * sc;
    float v = (float)Y[t] * sc + sh;
    Anext[t] = (bf16_t)(v > 0.0f ? v : 0.0f);
}

// ---------------------------------------------------------------- max over k -> cat buffer
__global__ void maxk_kernel(const bf16_t* __restrict__ A, bf16_t* __restrict__ cat,
                            int C, int co, long long total /* Q*C */) {
    long long t = (long long)blockIdx.x * blockDim.x + threadIdx.x;
    if (t >= total) return;
    int c = (int)(t % C);
    long long q = t / C;
    const bf16_t* p = A + (q * KNN) * (long long)C + c;
    float m = 0.0f;                      // post-ReLU values are >= 0
    #pragma unroll
    for (int kk = 0; kk < KNN; ++kk) {
        float v = (float)p[(long long)kk * C];
        m = v > m ? v : m;
    }
    cat[q * 512 + co + c] = (bf16_t)m;
}

// ---------------------------------------------------------------- final BN + ReLU + transpose to (B, 512, N) f32
__global__ void bn_out_kernel(const bf16_t* __restrict__ Y,
                              const float* __restrict__ gsum, const float* __restrict__ gsq,
                              const float* __restrict__ gamma, const float* __restrict__ beta,
                              float* __restrict__ out, int N, int C, float invCnt, long long total) {
    long long t = (long long)blockIdx.x * blockDim.x + threadIdx.x;
    if (t >= total) return;
    int o = (int)(t % C);
    long long row = t / C;           // row = b*N + n
    int b = (int)(row / N);
    int n = (int)(row % N);
    float mean = gsum[o] * invCnt;
    float var  = gsq[o] * invCnt - mean * mean;
    float sc = gamma[o] * rsqrtf(var + 1e-5f);
    float sh = beta[o] - mean * sc;
    float v = (float)Y[t] * sc + sh;
    out[((size_t)b * C + o) * N + n] = v > 0.0f ? v : 0.0f;
}

// ---------------------------------------------------------------- host
extern "C" void kernel_launch(void* const* d_in, const int* in_sizes, int n_in,
                              void* d_out, int out_size, void* d_ws, size_t ws_size,
                              hipStream_t stream) {
    const int B = 8, N = 2048;
    const long long P1 = (long long)B * N * KNN;   // 327680 neighbor rows
    const long long Q  = (long long)B * N;         // 16384 points

    const float* xf  = (const float*)d_in[0];
    const float* W1f = (const float*)d_in[2];
    const float* W2f = (const float*)d_in[3];
    const float* W3f = (const float*)d_in[4];
    const float* W4f = (const float*)d_in[5];
    const float* W5f = (const float*)d_in[6];
    const float* g1 = (const float*)d_in[7];  const float* b1 = (const float*)d_in[8];
    const float* g2 = (const float*)d_in[9];  const float* b2 = (const float*)d_in[10];
    const float* g3 = (const float*)d_in[11]; const float* b3 = (const float*)d_in[12];
    const float* g4 = (const float*)d_in[13]; const float* b4 = (const float*)d_in[14];
    const float* g5 = (const float*)d_in[15]; const float* b5 = (const float*)d_in[16];
    float* out = (float*)d_out;

    // ---- workspace carve-out (256B aligned)
    char* base = (char*)d_ws;
    size_t off = 0;
    auto carve = [&](size_t bytes) {
        char* p = base + off;
        off = (off + bytes + 255) & ~(size_t)255;
        return p;
    };
    int*    idxp  = (int*)   carve((size_t)P1 * 4);
    bf16_t* Wb1   = (bf16_t*)carve(64 * 32 * 2);
    bf16_t* Wb2   = (bf16_t*)carve(64 * 64 * 2);
    bf16_t* Wb3   = (bf16_t*)carve(128 * 64 * 2);
    bf16_t* Wb4   = (bf16_t*)carve(256 * 128 * 2);
    bf16_t* Wb5   = (bf16_t*)carve(512 * 512 * 2);
    float*  stats = (float*) carve(1024 * 4);       // [0:512)=sum, [512:1024)=sumsq
    bf16_t* Aping = (bf16_t*)carve((size_t)P1 * 256 * 2);
    bf16_t* Apong = (bf16_t*)carve((size_t)P1 * 256 * 2);
    bf16_t* Ybuf  = (bf16_t*)carve((size_t)P1 * 256 * 2);
    bf16_t* catb  = (bf16_t*)carve((size_t)Q * 512 * 2);
    float* gsum = stats;
    float* gsq  = stats + 512;

    // ---- weight conversion
    wconv_kernel<<<(64 * 32 + 255) / 256, 256, 0, stream>>>(W1f, Wb1, 64, 6, 32);
    wconv_kernel<<<(64 * 64 + 255) / 256, 256, 0, stream>>>(W2f, Wb2, 64, 64, 64);
    wconv_kernel<<<(128 * 64 + 255) / 256, 256, 0, stream>>>(W3f, Wb3, 128, 64, 64);
    wconv_kernel<<<(256 * 128 + 255) / 256, 256, 0, stream>>>(W4f, Wb4, 256, 128, 128);
    wconv_kernel<<<(512 * 512 + 255) / 256, 256, 0, stream>>>(W5f, Wb5, 512, 512, 512);

    // ---- kNN + graph features
    knn_kernel<<<(B * N) / 4, 128, 0, stream>>>(xf, idxp);
    feat_kernel<<<(unsigned)((P1 + 255) / 256), 256, 0, stream>>>(xf, idxp, Aping, P1);

    const float inv1 = 1.0f / (float)P1;
    const float inv5 = 1.0f / (float)Q;
    dim3 blk(256);

    // ---- layer 1: 32(pad6) -> 64
    zerof_kernel<<<1, 1024, 0, stream>>>(stats, 1024);
    wmma_mm_kernel<<<dim3((unsigned)(P1 / 128), 64 / 64), blk, 0, stream>>>(Aping, Wb1, Ybuf, gsum, gsq, 32, 64);
    bn_apply_kernel<<<(unsigned)((P1 * 64 + 255) / 256), blk, 0, stream>>>(Ybuf, gsum, gsq, g1, b1, Apong, 64, inv1, P1 * 64);
    maxk_kernel<<<(unsigned)((Q * 64 + 255) / 256), blk, 0, stream>>>(Apong, catb, 64, 0, Q * 64);

    // ---- layer 2: 64 -> 64
    zerof_kernel<<<1, 1024, 0, stream>>>(stats, 1024);
    wmma_mm_kernel<<<dim3((unsigned)(P1 / 128), 64 / 64), blk, 0, stream>>>(Apong, Wb2, Ybuf, gsum, gsq, 64, 64);
    bn_apply_kernel<<<(unsigned)((P1 * 64 + 255) / 256), blk, 0, stream>>>(Ybuf, gsum, gsq, g2, b2, Aping, 64, inv1, P1 * 64);
    maxk_kernel<<<(unsigned)((Q * 64 + 255) / 256), blk, 0, stream>>>(Aping, catb, 64, 64, Q * 64);

    // ---- layer 3: 64 -> 128
    zerof_kernel<<<1, 1024, 0, stream>>>(stats, 1024);
    wmma_mm_kernel<<<dim3((unsigned)(P1 / 128), 128 / 64), blk, 0, stream>>>(Aping, Wb3, Ybuf, gsum, gsq, 64, 128);
    bn_apply_kernel<<<(unsigned)((P1 * 128 + 255) / 256), blk, 0, stream>>>(Ybuf, gsum, gsq, g3, b3, Apong, 128, inv1, P1 * 128);
    maxk_kernel<<<(unsigned)((Q * 128 + 255) / 256), blk, 0, stream>>>(Apong, catb, 128, 128, Q * 128);

    // ---- layer 4: 128 -> 256
    zerof_kernel<<<1, 1024, 0, stream>>>(stats, 1024);
    wmma_mm_kernel<<<dim3((unsigned)(P1 / 128), 256 / 64), blk, 0, stream>>>(Apong, Wb4, Ybuf, gsum, gsq, 128, 256);
    bn_apply_kernel<<<(unsigned)((P1 * 256 + 255) / 256), blk, 0, stream>>>(Ybuf, gsum, gsq, g4, b4, Aping, 256, inv1, P1 * 256);
    maxk_kernel<<<(unsigned)((Q * 256 + 255) / 256), blk, 0, stream>>>(Aping, catb, 256, 256, Q * 256);

    // ---- layer 5: cat(512) -> 512, BN + ReLU + transpose store
    zerof_kernel<<<1, 1024, 0, stream>>>(stats, 1024);
    wmma_mm_kernel<<<dim3((unsigned)(Q / 128), 512 / 64), blk, 0, stream>>>(catb, Wb5, Ybuf, gsum, gsq, 512, 512);
    bn_out_kernel<<<(unsigned)((Q * 512 + 255) / 256), blk, 0, stream>>>(Ybuf, gsum, gsq, g5, b5, out, N, 512, inv5, Q * 512);
}
